// Mamba_84267258347738
// MI455X (gfx1250) — matrix operand order repaired and checked by reading, not compiled, not run on hardware
//
#include <hip/hip_runtime.h>

// ---------------------------------------------------------------------------
// Mamba forward for MI455X (gfx1250, wave32, WMMA).
// GEMMs: v_wmma_f32_16x16x32_f16 (f32 accumulate), double-buffered LDS tiles
// staged with CDNA5 async global->LDS DMA (ASYNCcnt software pipeline).
// Selective scan: one lane per (batch, channel), B/C chunks staged in LDS.
// ---------------------------------------------------------------------------

typedef __attribute__((ext_vector_type(16))) _Float16 v16h;
typedef __attribute__((ext_vector_type(8)))  _Float16 v8h;
typedef __attribute__((ext_vector_type(8)))  float    v8f;

#define BATCH   2
#define SEQLEN  1024
#define DMODEL  1024
#define DINNER  2048
#define DSTATE  16
#define DCONV   4
#define DTRANK  64
#define GDIM    96                     // DT_RANK + 2*D_STATE
#define MROWS   (BATCH * SEQLEN)       // 2048

// ---- CDNA5 async global->LDS copy (16 bytes), tracked by ASYNCcnt ----
__device__ __forceinline__ void async_copy_b128(unsigned lds_byte_off,
                                                const void* gaddr) {
  asm volatile("global_load_async_to_lds_b128 %0, %1, off"
               :: "v"(lds_byte_off), "v"(gaddr)
               : "memory");
}
// async loads complete in order: "<= 8" means the previous tile's 8 ops landed
__device__ __forceinline__ void wait_async_le8() {
  asm volatile("s_wait_asynccnt 0x8" ::: "memory");
}
__device__ __forceinline__ void wait_async_le0() {
  asm volatile("s_wait_asynccnt 0x0" ::: "memory");
}

// ---------------- f32 -> f16 conversion ----------------
__global__ void cvt_f16_kernel(const float* __restrict__ src,
                               _Float16* __restrict__ dst, long n) {
  long i = (long)blockIdx.x * blockDim.x + threadIdx.x;
  if (i < n) dst[i] = (_Float16)src[i];
}

// ---------------- WMMA NT GEMM: C[M,N] = A[M,K] * W[N,K]^T ----------------
// 128 threads = 4 waves. Block tile 64x64, K step 64 (2 wmma k-slices).
// Double-buffered LDS; tile k+1 is DMA'd (async->LDS) while tile k computes.
// Each wave owns a 16-row strip and issues 8 wmma per K step.
#define LDTW 72   // LDS row stride in halves (128B rows padded to 144B)
__global__ __launch_bounds__(128)
void gemm_nt_f16(const _Float16* __restrict__ A,
                 const _Float16* __restrict__ W,
                 float* __restrict__ C,
                 int M, int N, int K) {
  __shared__ _Float16 As[2][64 * LDTW];
  __shared__ _Float16 Ws[2][64 * LDTW];
  __shared__ alignas(64) _Float16 dump[32];   // sink for uniform-issue OOB loads

  const int m0   = blockIdx.y * 64;
  const int n0   = blockIdx.x * 64;
  const int tid  = threadIdx.x;
  const int wave = tid >> 5;
  const int lane = tid & 31;
  const int hi   = lane >> 4;   // half-wave select (ISA 16-bit fragment layout)
  const int mr   = lane & 15;

  // per-thread staging slot: one 64-byte half-row of each tile (4 x b128)
  const int row = tid >> 1, seg = tid & 1;
  const int gm = m0 + row;
  const int gn = n0 + row;
  const bool a_ok = (gm < M);
  const bool w_ok = (gn < N);
  // uniform issue: OOB rows load a clamped (valid) address into the dump slot
  const _Float16* gA = &A[(long)(a_ok ? gm : 0) * K + seg * 32];
  const _Float16* gW = &W[(long)(w_ok ? gn : 0) * K + seg * 32];
  const unsigned dumpoff = (unsigned)(size_t)&dump[0];
  const unsigned aoff[2] = {
    a_ok ? (unsigned)(size_t)&As[0][row * LDTW + seg * 32] : dumpoff,
    a_ok ? (unsigned)(size_t)&As[1][row * LDTW + seg * 32] : dumpoff };
  const unsigned woff[2] = {
    w_ok ? (unsigned)(size_t)&Ws[0][row * LDTW + seg * 32] : dumpoff,
    w_ok ? (unsigned)(size_t)&Ws[1][row * LDTW + seg * 32] : dumpoff };

  // OOB rows contribute zeros (written once; async ops never touch them)
  if (!a_ok) {
    v8h z = {};
    #pragma unroll
    for (int q = 0; q < 2; ++q)
      #pragma unroll
      for (int c = 0; c < 4; ++c)
        *(v8h*)&As[q][row * LDTW + seg * 32 + c * 8] = z;
  }
  if (!w_ok) {
    v8h z = {};
    #pragma unroll
    for (int q = 0; q < 2; ++q)
      #pragma unroll
      for (int c = 0; c < 4; ++c)
        *(v8h*)&Ws[q][row * LDTW + seg * 32 + c * 8] = z;
  }

  // stage tile starting at k0 into buffer p: 8 async b128 per thread
  auto stage = [&](int p, int k0) {
    const char* sa = (const char*)(gA + k0);
    const char* sw = (const char*)(gW + k0);
    #pragma unroll
    for (int c = 0; c < 4; ++c) async_copy_b128(aoff[p] + 16 * c, sa + 16 * c);
    #pragma unroll
    for (int c = 0; c < 4; ++c) async_copy_b128(woff[p] + 16 * c, sw + 16 * c);
  };

  v8f acc[4] = {};

  stage(0, 0);
  int p = 0;
  for (int k0 = 0; k0 < K; k0 += 64, p ^= 1) {
    if (k0 + 64 < K) { stage(p ^ 1, k0 + 64); wait_async_le8(); }
    else             { wait_async_le0(); }
    __syncthreads();   // tile k landed in all waves; prior reads of buf p done

    const _Float16* as = &As[p][0];
    const _Float16* ws = &Ws[p][0];
    #pragma unroll
    for (int s = 0; s < 2; ++s) {
      // A fragment: ISA 16-bit A layout:
      // lane (mr,hi) element j holds K = s*32 + (j&7) + (j>>3)*16 + hi*8
      v16h afrag;
      #pragma unroll
      for (int j = 0; j < 16; ++j) {
        const int kk = s * 32 + (j & 7) + ((j >> 3) << 4) + (hi << 3);
        afrag[j] = as[(wave * 16 + mr) * LDTW + kk];
      }
      // B fragments: lane (mr,hi) element e holds K = s*32 + (e&7) + (e>>3)*16 + hi*8, N = mr
      #pragma unroll
      for (int jt = 0; jt < 4; ++jt) {
        v16h bfrag;
        #pragma unroll
        for (int e = 0; e < 16; ++e)
          bfrag[e] = ws[(jt * 16 + mr) * LDTW + s * 32 + (e & 7) + ((e >> 3) << 4) + (hi << 3)];
        acc[jt] = __builtin_amdgcn_wmma_f32_16x16x32_f16(
            false, afrag, false, bfrag, (short)0, acc[jt], false, false);
      }
    }
    __syncthreads();   // reads of buf p done before it is re-staged
  }

  // ---- store: f32 C/D layout: VGPR r -> M = r + hi*8, N = mr ----
  #pragma unroll
  for (int jt = 0; jt < 4; ++jt) {
    #pragma unroll
    for (int r = 0; r < 8; ++r) {
      const int m = m0 + wave * 16 + r + hi * 8;
      const int n = n0 + jt * 16 + mr;
      if (m < M && n < N) C[(long)m * N + n] = acc[jt][r];
    }
  }
}

// ---------------- causal depthwise conv (k=4) + SiLU ----------------
// x lives in xz[:, :, 0:DINNER]; emits xc in f32 (for scan) and f16 (for x_proj GEMM)
__global__ void conv_silu_kernel(const float* __restrict__ xz,
                                 const float* __restrict__ cw,
                                 const float* __restrict__ cb,
                                 float* __restrict__ xc32,
                                 _Float16* __restrict__ xc16) {
  const long i = (long)blockIdx.x * blockDim.x + threadIdx.x;  // over B*L*DINNER
  if (i >= (long)BATCH * SEQLEN * DINNER) return;
  const int  d    = (int)(i % DINNER);
  const long bl   = i / DINNER;
  const int  l    = (int)(bl % SEQLEN);
  const long brow = bl - l;                                    // b*SEQLEN
  float acc = cb[d];
  #pragma unroll
  for (int k = 0; k < DCONV; ++k) {
    const int ll = l - (DCONV - 1) + k;
    if (ll >= 0)
      acc += cw[d * DCONV + k] * xz[(brow + ll) * (long)(2 * DINNER) + d];
  }
  const float s = acc / (1.f + __expf(-acc));   // SiLU
  xc32[i] = s;
  xc16[i] = (_Float16)s;
}

// ---------------- text gate: sigmoid(x_text @ gate_w^T + gate_b) ----------------
__global__ __launch_bounds__(128)
void gate_kernel(const float* __restrict__ x_text,
                 const float* __restrict__ gate_w,
                 const float* __restrict__ gate_b,
                 float* __restrict__ gate) {
  __shared__ float red[128];
  const int o = blockIdx.x;           // 0 .. BATCH*GDIM-1
  const int b = o / GDIM, g = o % GDIM;
  float s = 0.f;
  for (int k = threadIdx.x; k < DMODEL; k += 128)
    s += x_text[b * DMODEL + k] * gate_w[g * DMODEL + k];
  red[threadIdx.x] = s;
  __syncthreads();
  for (int st = 64; st > 0; st >>= 1) {
    if (threadIdx.x < st) red[threadIdx.x] += red[threadIdx.x + st];
    __syncthreads();
  }
  if (threadIdx.x == 0)
    gate[o] = 1.f / (1.f + __expf(-(red[0] + gate_b[g])));
}

// ---------------- x_dbl *= gate (broadcast over L); extract dt (f16) ----------------
__global__ void modulate_kernel(float* __restrict__ xdbl,
                                const float* __restrict__ gate,
                                _Float16* __restrict__ dt16) {
  const int i = blockIdx.x * blockDim.x + threadIdx.x;   // over MROWS*GDIM
  if (i >= MROWS * GDIM) return;
  const int g = i % GDIM;
  const int m = i / GDIM;
  const int b = m / SEQLEN;
  const float v = xdbl[i] * gate[b * GDIM + g];
  xdbl[i] = v;
  if (g < DTRANK) dt16[m * DTRANK + g] = (_Float16)v;
}

// ---------------- delta = softplus(raw + dt_proj_b) ----------------
__global__ void softplus_kernel(float* __restrict__ delta,
                                const float* __restrict__ bias) {
  const long i = (long)blockIdx.x * blockDim.x + threadIdx.x;  // over MROWS*DINNER
  if (i >= (long)MROWS * DINNER) return;
  const int d = (int)(i % DINNER);
  const float x = delta[i] + bias[d];
  delta[i] = (x > 20.f) ? x : log1pf(__expf(x));
}

// ---------------- selective scan, fused with D*xc skip + SiLU(z) gate ----------------
// One lane per (b, d): 16-state recurrence in registers; B/C (shared across all
// channels of a batch) staged into LDS in 64-step chunks (two barriers / chunk).
#define CHUNK 64
__global__ __launch_bounds__(256)
void scan_kernel(const float* __restrict__ delta,
                 const float* __restrict__ xc,
                 const float* __restrict__ xdbl,
                 const float* __restrict__ xz,
                 const float* __restrict__ A_log,
                 const float* __restrict__ Dp,
                 _Float16* __restrict__ yact) {
  __shared__ float sB[CHUNK][DSTATE];
  __shared__ float sC[CHUNK][DSTATE];
  const int c = blockIdx.x * 256 + threadIdx.x;   // global channel; block never straddles a batch
  const int b = c / DINNER;
  const int d = c % DINNER;

  float a[DSTATE], h[DSTATE];
  #pragma unroll
  for (int n = 0; n < DSTATE; ++n) {
    a[n] = -__expf(A_log[d * DSTATE + n]);
    h[n] = 0.f;
  }
  const float Dv = Dp[d];

  for (int t0 = 0; t0 < SEQLEN; t0 += CHUNK) {
    __syncthreads();
    for (int f = threadIdx.x; f < CHUNK * 32; f += 256) {
      const int row = f >> 5, col = f & 31;
      const float v = xdbl[(long)(b * SEQLEN + t0 + row) * GDIM + DTRANK + col];
      if (col < DSTATE) sB[row][col] = v;
      else              sC[row][col - DSTATE] = v;
    }
    __syncthreads();
    for (int tt = 0; tt < CHUNK; ++tt) {
      const long idx = (long)(b * SEQLEN + t0 + tt) * DINNER + d;
      const float dv = delta[idx];
      const float xv = xc[idx];
      const float dx = dv * xv;
      float y = 0.f;
      #pragma unroll
      for (int n = 0; n < DSTATE; ++n) {
        h[n] = h[n] * __expf(dv * a[n]) + dx * sB[tt][n];
        y += h[n] * sC[tt][n];
      }
      const float z = xz[(long)(b * SEQLEN + t0 + tt) * (2 * DINNER) + DINNER + d];
      const float g = z / (1.f + __expf(-z));    // SiLU(z)
      yact[idx] = (_Float16)((y + Dv * xv) * g);
    }
  }
}

// ---------------------------------------------------------------------------
// Host-side orchestration
// ---------------------------------------------------------------------------
static inline size_t align256(size_t x) { return (x + 255) & ~(size_t)255; }

extern "C" void kernel_launch(void* const* d_in, const int* in_sizes, int n_in,
                              void* d_out, int out_size, void* d_ws, size_t ws_size,
                              hipStream_t stream) {
  const float* hs        = (const float*)d_in[0];   // (B,L,DMODEL)
  const float* x_text    = (const float*)d_in[1];   // (B,DMODEL)
  const float* in_proj_w = (const float*)d_in[2];   // (2*DINNER, DMODEL)
  const float* conv_w    = (const float*)d_in[3];   // (DINNER,1,DCONV)
  const float* conv_b    = (const float*)d_in[4];   // (DINNER)
  const float* x_proj_w  = (const float*)d_in[5];   // (GDIM, DINNER)
  const float* dt_proj_w = (const float*)d_in[6];   // (DINNER, DTRANK)
  const float* dt_proj_b = (const float*)d_in[7];   // (DINNER)
  const float* A_log     = (const float*)d_in[8];   // (DINNER, DSTATE)
  const float* Dp        = (const float*)d_in[9];   // (DINNER)
  const float* out_proj_w= (const float*)d_in[10];  // (DMODEL, DINNER)
  const float* gate_w    = (const float*)d_in[11];  // (GDIM, DMODEL)
  const float* gate_b    = (const float*)d_in[12];  // (GDIM)
  float* out = (float*)d_out;                       // (B,L,DMODEL)

  // ---- workspace carve-up ----
  char* p = (char*)d_ws;
  size_t off = 0;
  auto take = [&](size_t bytes) { char* q = p + off; off = align256(off + bytes); return q; };

  _Float16* hs16     = (_Float16*)take((size_t)MROWS * DMODEL * 2);
  _Float16* w_in16   = (_Float16*)take((size_t)2 * DINNER * DMODEL * 2);
  _Float16* w_xp16   = (_Float16*)take((size_t)GDIM * DINNER * 2);
  _Float16* w_dt16   = (_Float16*)take((size_t)DINNER * DTRANK * 2);
  _Float16* w_out16  = (_Float16*)take((size_t)DMODEL * DINNER * 2);
  float*    xz       = (float*)   take((size_t)MROWS * 2 * DINNER * 4);
  float*    xc32     = (float*)   take((size_t)MROWS * DINNER * 4);
  _Float16* xc16     = (_Float16*)take((size_t)MROWS * DINNER * 2);
  float*    xdbl     = (float*)   take((size_t)MROWS * GDIM * 4);
  float*    gatev    = (float*)   take((size_t)BATCH * GDIM * 4);
  _Float16* dt16     = (_Float16*)take((size_t)MROWS * DTRANK * 2);
  float*    delta    = (float*)   take((size_t)MROWS * DINNER * 4);
  _Float16* yact16   = (_Float16*)take((size_t)MROWS * DINNER * 2);
  (void)ws_size; (void)n_in; (void)in_sizes; (void)out_size;

  // ---- 1. f32 -> f16 conversions (activations + weights) ----
  auto cvt = [&](const float* s, _Float16* d, long n) {
    cvt_f16_kernel<<<(unsigned)((n + 255) / 256), 256, 0, stream>>>(s, d, n);
  };
  cvt(hs,         hs16,    (long)MROWS * DMODEL);
  cvt(in_proj_w,  w_in16,  (long)2 * DINNER * DMODEL);
  cvt(x_proj_w,   w_xp16,  (long)GDIM * DINNER);
  cvt(dt_proj_w,  w_dt16,  (long)DINNER * DTRANK);
  cvt(out_proj_w, w_out16, (long)DMODEL * DINNER);

  // ---- 2. in_proj: xz = hs @ in_proj_w^T   (2048 x 4096 x K1024) ----
  gemm_nt_f16<<<dim3(4 * DINNER / 64, MROWS / 64), 128, 0, stream>>>(
      hs16, w_in16, xz, MROWS, 2 * DINNER, DMODEL);

  // ---- 3. causal depthwise conv + SiLU ----
  conv_silu_kernel<<<(unsigned)(((long)MROWS * DINNER + 255) / 256), 256, 0, stream>>>(
      xz, conv_w, conv_b, xc32, xc16);

  // ---- 4. x_proj: x_dbl = xc @ x_proj_w^T   (2048 x 96 x K2048) ----
  gemm_nt_f16<<<dim3((GDIM + 63) / 64, MROWS / 64), 128, 0, stream>>>(
      xc16, w_xp16, xdbl, MROWS, GDIM, DINNER);

  // ---- 5. text gate ----
  gate_kernel<<<BATCH * GDIM, 128, 0, stream>>>(x_text, gate_w, gate_b, gatev);

  // ---- 6. gate modulation + dt extraction ----
  modulate_kernel<<<(MROWS * GDIM + 255) / 256, 256, 0, stream>>>(xdbl, gatev, dt16);

  // ---- 7. dt_proj: delta_raw = dt @ dt_proj_w^T   (2048 x 2048 x K64) ----
  gemm_nt_f16<<<dim3(DINNER / 64, MROWS / 64), 128, 0, stream>>>(
      dt16, w_dt16, delta, MROWS, DINNER, DTRANK);

  // ---- 8. delta = softplus(raw + bias) ----
  softplus_kernel<<<(unsigned)(((long)MROWS * DINNER + 255) / 256), 256, 0, stream>>>(
      delta, dt_proj_b);

  // ---- 9. selective scan (fused skip + SiLU(z) gate, emits f16) ----
  scan_kernel<<<BATCH * DINNER / 256, 256, 0, stream>>>(
      delta, xc32, xdbl, xz, A_log, Dp, yact16);

  // ---- 10. out_proj: out = yact @ out_proj_w^T   (2048 x 1024 x K2048) ----
  gemm_nt_f16<<<dim3(DMODEL / 64, MROWS / 64), 128, 0, stream>>>(
      yact16, w_out16, out, MROWS, DMODEL, DINNER);
}
